// EKPConv_v1_35725537968373
// MI455X (gfx1250) — matrix operand order repaired and checked by reading, not compiled, not run on hardware
//
#include <hip/hip_runtime.h>

typedef __attribute__((ext_vector_type(16))) _Float16 v16h;
typedef __attribute__((ext_vector_type(8)))  _Float16 v8h;
typedef __attribute__((ext_vector_type(8)))  float    v8f;

#define KPTS   15
#define IN_CH  64
#define OUT_CH 128
#define N_LRF  4
#define HN     32
#define QT     16          // queries per workgroup tile
#define KC_PAD 1024        // padded contraction: kc = l*256 + c*16 + k, k in [0,16)
#define NSTEP  32          // 1024 / 32
#define KP_INV (1.0f / 1.2f)

union FragH  { uint4 u[2]; v16h h; };
union PackH  { v8h h; uint4 u; };

// ---------------------------------------------------------------------------
// Pre-pack weights [K, IN_CH, OUT_CH] f32 -> f16 pairs in WMMA B-fragment
// order over the PERMUTED padded contraction index kc = l*256 + c*16 + k
// (k = 15 slots are zero-filled; the matching wf slots are also zero).
// dst[((t*NSTEP+s)*32 + lane)*8 + v] = pair (kc, kc+1) for o = t*16 + lane%16,
// kc = s*32 + (lane/16)*16 + 2v.  256 KB total -> L2 resident, reused by WGs.
// ---------------------------------------------------------------------------
__global__ void ekp_pack_weights(const float* __restrict__ w,
                                 unsigned* __restrict__ dst) {
  int id = blockIdx.x * blockDim.x + threadIdx.x;
  if (id >= 8 * NSTEP * 32 * 8) return;
  int v    = id & 7;
  int lane = (id >> 3) & 31;
  int ts   = id >> 8;
  int s    = ts & (NSTEP - 1);
  int t    = ts >> 5;
  int o    = t * 16 + (lane & 15);
  int kc   = s * 32 + ((lane >> 4) << 4) + 2 * v;   // low 4 bits = k (even)
  int l    = kc >> 8;
  int c    = (kc >> 4) & 15;
  int k0   = kc & 15;                               // even, so k0+1 <= 15
  float f0 = (k0 < KPTS)     ? w[(size_t)(k0 * IN_CH + l * 16 + c) * OUT_CH + o] : 0.f;
  float f1 = (k0 + 1 < KPTS) ? w[(size_t)((k0 + 1) * IN_CH + l * 16 + c) * OUT_CH + o] : 0.f;
  _Float16 lo = (_Float16)f0, hi = (_Float16)f1;
  unsigned ulo = (unsigned)__builtin_bit_cast(unsigned short, lo);
  unsigned uhi = (unsigned)__builtin_bit_cast(unsigned short, hi);
  dst[id] = ulo | (uhi << 16);
}

// ---------------------------------------------------------------------------
// Fused EKPConv: one workgroup (8 wave32) handles 16 query points.
// LDS = 64KB Winf + 64KB featT + 32KB wf = 163840 B -> exactly 2 WGs per WGP.
// ---------------------------------------------------------------------------
__global__ void __launch_bounds__(256) ekp_conv(
    const float* __restrict__ q_pts, const float* __restrict__ s_pts,
    const int*   __restrict__ nbidx, const float* __restrict__ x,
    const float* __restrict__ lrf,   const float* __restrict__ kp,
    const unsigned* __restrict__ wpack, float* __restrict__ out, int M) {
  extern __shared__ unsigned char smem_raw[];
  _Float16* Winf  = (_Float16*)smem_raw;   // [16 n][4 l][16 k(pad)][32 h] : A frags
  _Float16* featT = Winf + 32768;          // [16 n][4 l][16 c][32 h]      : B frags
  _Float16* wfL   = featT + 32768;         // [16 n][1024 kc]  kc = l*256+c*16+k

  const int tid  = threadIdx.x;
  const int lane = tid & 31;
  const int wlid = __builtin_amdgcn_readfirstlane(tid) >> 5;  // scalar wave id
  const int n0   = blockIdx.x * QT;

  // ---- Stage 1: gather features + kernel-point influence weights ----------
  // Whole wave shares one query point n -> q_pts / lrf become scalar loads.
  #pragma unroll
  for (int pp = 0; pp < 2; ++pp) {
    const int n   = pp * 8 + wlid;         // scalar
    const int h   = lane;
    const int gq  = n0 + n;                // scalar
    const int idx = nbidx[(size_t)gq * HN + h];
    const bool valid = (idx < M);

    float sx, sy, sz;
    if (valid) {
      sx = s_pts[(size_t)idx * 3 + 0];
      sy = s_pts[(size_t)idx * 3 + 1];
      sz = s_pts[(size_t)idx * 3 + 2];
    } else {
      sx = sy = sz = 1.0e6f;               // shadow point, matches reference
    }
    const float d0 = sx - q_pts[(size_t)gq * 3 + 0];   // scalar q load
    const float d1 = sy - q_pts[(size_t)gq * 3 + 1];
    const float d2 = sz - q_pts[(size_t)gq * 3 + 2];

    // Gather neighbor features into B-fragment order [n][l*16+c][h].
    {
      const unsigned fb = (unsigned)n * (IN_CH * HN) + h;
      const float4* xr = (const float4*)(x + (size_t)(valid ? idx : 0) * IN_CH);
      #pragma unroll
      for (int cc = 0; cc < 16; ++cc) {
        float4 f;
        if (valid) f = xr[cc];
        else       f = make_float4(0.f, 0.f, 0.f, 0.f);
        const int c4 = cc * 4;
        featT[fb + (unsigned)(c4 + 0) * HN] = (_Float16)f.x;
        featT[fb + (unsigned)(c4 + 1) * HN] = (_Float16)f.y;
        featT[fb + (unsigned)(c4 + 2) * HN] = (_Float16)f.z;
        featT[fb + (unsigned)(c4 + 3) * HN] = (_Float16)f.w;
      }
    }

    // Influence weights into A-fragment order [n][l][k(pad16)][h].
    #pragma unroll
    for (int l = 0; l < N_LRF; ++l) {
      const float* Lm = lrf + ((size_t)gq * N_LRF + l) * 9;  // scalar loads
      const float a0 = d0 * Lm[0] + d1 * Lm[3] + d2 * Lm[6];
      const float a1 = d0 * Lm[1] + d1 * Lm[4] + d2 * Lm[7];
      const float a2 = d0 * Lm[2] + d1 * Lm[5] + d2 * Lm[8];
      const unsigned wb = (unsigned)((n * N_LRF + l) * 16) * HN + h;
      #pragma unroll
      for (int k = 0; k < KPTS; ++k) {
        const float e0 = a0 - kp[k * 3 + 0];            // scalar kp loads
        const float e1 = a1 - kp[k * 3 + 1];
        const float e2 = a2 - kp[k * 3 + 2];
        float wgt = 1.0f - sqrtf(e0 * e0 + e1 * e1 + e2 * e2) * KP_INV;
        wgt = fmaxf(wgt, 0.0f);
        Winf[wb + (unsigned)k * HN] = (_Float16)wgt;
      }
      Winf[wb + 15u * HN] = (_Float16)0.0f;   // padded k row -> wf pad = 0
    }
  }
  __syncthreads();

  const int lr = lane & 15;
  const int lh = lane >> 4;

  // ---- Stage 2: wf[n][l*256 + c*16 + k] = Σ_h Winf[k,h] * feat[h,c] -------
  // Two-deep software pipeline: load frags for pair pp+1, WMMA pair pp,
  // convert+store pair pp-1 (fills the WMMA->VALU hazard slots with work).
  {
    auto ld2 = [&](int pr, FragH& A, FragH& B) {
      const int n = pr >> 2, l = pr & 3;
      const _Float16* ar =
          Winf + (unsigned)((n * N_LRF + l) * 16 + lr) * HN + lh * 8;
      A.u[0] = *(const uint4*)(ar);        // K(=h) = hb..hb+7
      A.u[1] = *(const uint4*)(ar + 16);   // K(=h) = hb+16..hb+23
      const _Float16* br =
          featT + (unsigned)((n * N_LRF + l) * 16 + lr) * HN + lh * 8;
      B.u[0] = *(const uint4*)(br);        // K(=h) = hb..hb+7
      B.u[1] = *(const uint4*)(br + 16);   // K(=h) = hb+16..hb+23
    };
    auto st2 = [&](int pr, const v8f& d) {
      // D: lane holds c = lr, rows k = lh*8 + r -> contiguous in wf layout.
      const int n = pr >> 2, l = pr & 3;
      PackH st;
      #pragma unroll
      for (int r = 0; r < 8; ++r) st.h[r] = (_Float16)d[r];
      *(uint4*)(wfL + (unsigned)n * KC_PAD +
                (unsigned)(l * 256 + lr * 16) + lh * 8) = st.u;
    };

    FragH A2[2], B2[2];
    v8f d2[2];
    ld2(wlid * 8, A2[0], B2[0]);
    #pragma unroll
    for (int pp = 0; pp < 8; ++pp) {
      const int cur = pp & 1;
      if (pp + 1 < 8) ld2(wlid * 8 + pp + 1, A2[cur ^ 1], B2[cur ^ 1]);
      v8f z = {};
      d2[cur] = __builtin_amdgcn_wmma_f32_16x16x32_f16(
          false, A2[cur].h, false, B2[cur].h, (short)0, z, false, false);
      if (pp > 0) st2(wlid * 8 + pp - 1, d2[cur ^ 1]);
    }
    st2(wlid * 8 + 7, d2[1]);
  }
  __syncthreads();

  // ---- Stage 3: out[16 x 128] = wf[16 x 1024] x Wpack[1024 x 128] ---------
  // Double-buffered: A from LDS (DScnt), B from global (LOADcnt) -> next
  // step's loads overlap the current WMMA on independent counters.
  {
    auto ld3 = [&](int s, FragH& A, FragH& B) {
      const _Float16* ar = wfL + (unsigned)lr * KC_PAD + s * 32 + lh * 16;
      A.u[0] = *(const uint4*)(ar);
      A.u[1] = *(const uint4*)(ar + 8);
      const uint4* bp = (const uint4*)(wpack +
                         (((unsigned)wlid * NSTEP + s) * 32 + lane) * 8);
      B.u[0] = bp[0];
      B.u[1] = bp[1];
    };

    v8f acc = {};
    FragH A3[2], B3[2];
    ld3(0, A3[0], B3[0]);
    #pragma unroll 2
    for (int s = 0; s < NSTEP - 1; ++s) {
      const int cur = s & 1;
      ld3(s + 1, A3[cur ^ 1], B3[cur ^ 1]);
      acc = __builtin_amdgcn_wmma_f32_16x16x32_f16(
          false, A3[cur].h, false, B3[cur].h, (short)0, acc, false, false);
    }
    acc = __builtin_amdgcn_wmma_f32_16x16x32_f16(
        false, A3[1].h, false, B3[1].h, (short)0, acc, false, false);

    float* op = out + (size_t)n0 * OUT_CH + wlid * 16 + lr;
    #pragma unroll
    for (int r = 0; r < 8; ++r) {
      op[(size_t)(r + lh * 8) * OUT_CH] = acc[r];
    }
  }
}

extern "C" void kernel_launch(void* const* d_in, const int* in_sizes, int n_in,
                              void* d_out, int out_size, void* d_ws, size_t ws_size,
                              hipStream_t stream) {
  const float* q_pts = (const float*)d_in[0];
  const float* s_pts = (const float*)d_in[1];
  const int*   nbidx = (const int*)d_in[2];
  const float* x     = (const float*)d_in[3];
  const float* lrf   = (const float*)d_in[4];
  const float* wts   = (const float*)d_in[5];
  const float* kp    = (const float*)d_in[6];
  float* out = (float*)d_out;
  unsigned* wpack = (unsigned*)d_ws;     // needs 65536 dwords = 256 KB

  const int N = in_sizes[0] / 3;
  const int M = in_sizes[1] / 3;

  const int packDwords = 8 * NSTEP * 32 * 8;   // 65536
  ekp_pack_weights<<<(packDwords + 255) / 256, 256, 0, stream>>>(wts, wpack);

  const size_t smem = (size_t)(32768 + 32768 + 16384) * sizeof(_Float16); // 163840
  (void)hipFuncSetAttribute((const void*)ekp_conv,
                            hipFuncAttributeMaxDynamicSharedMemorySize,
                            (int)smem);
  ekp_conv<<<N / QT, 256, smem, stream>>>(q_pts, s_pts, nbidx, x, lrf, kp,
                                          wpack, out, M);
}
